// DCNv2_47115791237771
// MI455X (gfx1250) — matrix operand order repaired and checked by reading, not compiled, not run on hardware
//
#include <hip/hip_runtime.h>
#include <hip/hip_bf16.h>

// DCNv2 fused deformable-im2col + WMMA GEMM for gfx1250 (MI455X).
// Phase 0: pack weights f32 -> f16, transposed to [tap][cinChunk][cout][32]
//          so the GEMM kernel's A-tile fetches are coalesced b128.
// Phase 1: one workgroup per (n, ho): computes out[n, 0:256, ho, 0:64].
//          Flat pipeline over 9 taps x 8 cin-chunks of 32:
//            - A-tile (256x32 f16) fetched with ASYNC global->LDS b128 into a
//              double buffer, issued one iteration ahead (hidden behind WMMA),
//            - B-tile (32x64 f16) built by bilinear gather (4 loads/sample,
//              coords+mask precomputed per tap),
//            - 8 waves each run 2x4 v_wmma_f32_16x16x32_f16, f32 accumulate.

typedef __attribute__((ext_vector_type(16))) _Float16 v16h;
typedef __attribute__((ext_vector_type(8)))  float    v8f;
typedef int v4i __attribute__((vector_size(16)));   // int4 as expected by builtin

#define CIN   256
#define COUT  256
#define HH    64
#define WW    64
#define KTAPS 9
#define HW    (HH * WW)
#define NITER (KTAPS * 8)   // 72 (tap, cin-chunk) steps

#if defined(__gfx1250__) && __has_builtin(__builtin_amdgcn_global_load_async_to_lds_b128)
#define HAS_ASYNC 1
#else
#define HAS_ASYNC 0
#endif

__device__ __forceinline__ int iclamp(int v, int lo, int hi) {
    return v < lo ? lo : (v > hi ? hi : v);
}

#if HAS_ASYNC
#define GAS __attribute__((address_space(1)))
#define LAS __attribute__((address_space(3)))
__device__ __forceinline__ void async_ld_b128(const void* g, void* l) {
    // prototype (from clang diagnostic): (global int4*, lds int4*, imm offset, imm cpol)
    __builtin_amdgcn_global_load_async_to_lds_b128((GAS v4i*)g, (LAS v4i*)l, 0, 0);
}
#endif

__device__ __forceinline__ void wait_async(int n) {
#if HAS_ASYNC
    if (n == 0) {
#if __has_builtin(__builtin_amdgcn_s_wait_asynccnt)
        __builtin_amdgcn_s_wait_asynccnt(0);
#else
        asm volatile("s_wait_asynccnt 0x0" ::: "memory");
#endif
    } else {
#if __has_builtin(__builtin_amdgcn_s_wait_asynccnt)
        __builtin_amdgcn_s_wait_asynccnt(1);
#else
        asm volatile("s_wait_asynccnt 0x1" ::: "memory");
#endif
    }
#else
    (void)n;
#endif
}

// ---------------------------------------------------------------------------
// Weight pack: wf16[((k*8+cc)*256 + o)*32 + j] = (f16) w[(o*256 + cc*32+j)*9 + k]
// total elements = 9*8*256*32 = 589824
// ---------------------------------------------------------------------------
__global__ void dcn_pack_weight(const float* __restrict__ w,
                                _Float16* __restrict__ wf16) {
    int idx = blockIdx.x * 256 + threadIdx.x;      // < 589824
    int j  = idx & 31;
    int o  = (idx >> 5) & 255;
    int cc = (idx >> 13) & 7;
    int k  = idx >> 16;                            // 0..8
    wf16[idx] = (_Float16)w[((size_t)o * CIN + cc * 32 + j) * KTAPS + k];
}

// ---------------------------------------------------------------------------
// Fused deformable conv kernel
// ---------------------------------------------------------------------------
__global__ __launch_bounds__(256)
void dcn_fused(const float* __restrict__ input,
               const float* __restrict__ offset,
               const float* __restrict__ maskp,
               const _Float16* __restrict__ wf16,
               const float* __restrict__ bias,
               float* __restrict__ out) {
    __shared__ __align__(16) _Float16 sA[2][COUT * 32]; // A tiles (double buffer)
    __shared__ __align__(16) _Float16 sB[64 * 32];      // B tile: [pixel][cin_local]
    __shared__ int   sIdx[64][4];                       // bilinear corner indices
    __shared__ float sWt [64][4];                       // corner weights (mask folded)

    const int ho   = blockIdx.x;       // 0..63
    const int n    = blockIdx.y;       // 0..7
    const int tid  = threadIdx.x;      // 0..255
    const int lane = tid & 31;
    const int wave = tid >> 5;         // 0..7
    const int rowl = lane & 15;
    const int half = lane >> 4;        // 0/1
    const int wbase = wave * 32;       // cout base for this wave

    const unsigned* sBu = (const unsigned*)sB;

    v8f acc[2][4];
    #pragma unroll
    for (int mt = 0; mt < 2; ++mt)
        #pragma unroll
        for (int nt = 0; nt < 4; ++nt)
            acc[mt][nt] = (v8f){0.f, 0.f, 0.f, 0.f, 0.f, 0.f, 0.f, 0.f};

#if HAS_ASYNC
    {   // prologue: async-fetch A tile for iteration 0 into buffer 0
        const _Float16* src = wf16 + ((size_t)tid) * 32;   // (k=0, cc=0)
        _Float16* dst = &sA[0][tid * 32];
        async_ld_b128(src,      dst);
        async_ld_b128(src +  8, dst +  8);
        async_ld_b128(src + 16, dst + 16);
        async_ld_b128(src + 24, dst + 24);
    }
#endif

    for (int it = 0; it < NITER; ++it) {
        const int k  = it >> 3;
        const int cc = it & 7;

        // ---- per-tap bilinear coordinate precompute (64 pixels) ----
        if (cc == 0) {
            if (tid < 64) {
                const int wo  = tid;
                const int pix = ho * WW + wo;
                float dy = offset[((size_t)n * 18 + 2 * k    ) * HW + pix];
                float dx = offset[((size_t)n * 18 + 2 * k + 1) * HW + pix];
                float mv = maskp [((size_t)n *  9 +     k    ) * HW + pix];
                float y = (float)(ho - 1 + (k / 3)) + dy;   // stride=1,pad=1,dil=1
                float x = (float)(wo - 1 + (k % 3)) + dx;
                float y0f = floorf(y), x0f = floorf(x);
                float ly = y - y0f,    lx = x - x0f;
                int y0 = (int)y0f, x0 = (int)x0f;
                int y1 = y0 + 1,   x1 = x0 + 1;
                int y0c = iclamp(y0, 0, HH - 1), y1c = iclamp(y1, 0, HH - 1);
                int x0c = iclamp(x0, 0, WW - 1), x1c = iclamp(x1, 0, WW - 1);
                float vy0 = (y0 >= 0 && y0 < HH) ? 1.f : 0.f;
                float vy1 = (y1 >= 0 && y1 < HH) ? 1.f : 0.f;
                float vx0 = (x0 >= 0 && x0 < WW) ? 1.f : 0.f;
                float vx1 = (x1 >= 0 && x1 < WW) ? 1.f : 0.f;
                sIdx[wo][0] = y0c * WW + x0c;  sWt[wo][0] = (1.f-ly)*(1.f-lx)*vy0*vx0*mv;
                sIdx[wo][1] = y0c * WW + x1c;  sWt[wo][1] = (1.f-ly)*lx      *vy0*vx1*mv;
                sIdx[wo][2] = y1c * WW + x0c;  sWt[wo][2] = ly*(1.f-lx)      *vy1*vx0*mv;
                sIdx[wo][3] = y1c * WW + x1c;  sWt[wo][3] = ly*lx            *vy1*vx1*mv;
            }
            __syncthreads();   // publish coords before B-tile build
        }

#if HAS_ASYNC
        // ---- issue async A-tile fetch for next iteration (other buffer) ----
        if (it + 1 < NITER) {
            const _Float16* src = wf16 + (((size_t)(it + 1) * COUT) + tid) * 32;
            _Float16* dst = &sA[(it + 1) & 1][tid * 32];
            async_ld_b128(src,      dst);
            async_ld_b128(src +  8, dst +  8);
            async_ld_b128(src + 16, dst + 16);
            async_ld_b128(src + 24, dst + 24);
        }
#else
        // ---- fallback: synchronous A-tile copy for this iteration ----
        {
            const float4* src = (const float4*)(wf16 + (((size_t)it * COUT) + tid) * 32);
            float4* dst = (float4*)(&sA[it & 1][tid * 32]);
            dst[0] = src[0]; dst[1] = src[1]; dst[2] = src[2]; dst[3] = src[3];
        }
#endif

        // ---- B tile: 32 cins x 64 pixels, 8 bilinear samples per thread ----
        {
            const int cl = tid >> 3;            // cin_local 0..31
            const int p0 = (tid & 7) * 8;       // pixel base
            const float* base = input + ((size_t)n * CIN + cc * 32 + cl) * HW;
            #pragma unroll
            for (int i = 0; i < 8; ++i) {
                const int p = p0 + i;
                float v = sWt[p][0] * base[sIdx[p][0]]
                        + sWt[p][1] * base[sIdx[p][1]]
                        + sWt[p][2] * base[sIdx[p][2]]
                        + sWt[p][3] * base[sIdx[p][3]];
                sB[p * 32 + cl] = (_Float16)v;
            }
        }

        // current A buffer must be resident (keep next-iter fetch in flight)
        wait_async((it + 1 < NITER) ? 1 : 0);
        __syncthreads();

        // ---- WMMA: each wave does 2 (M) x 4 (N) 16x16x32 f16 MACs ----
        {
            const unsigned* sAu = (const unsigned*)sA[it & 1];
            union U { v16h v; unsigned u[8]; };
            U b[4];
            #pragma unroll
            for (int nt = 0; nt < 4; ++nt) {
                // B (32x16 KxN) wave32 layout: lane&15 -> N; dword d holds
                // K-pair index p = (d&3) + half*4 + (d>>2)*8 (same as A)
                const unsigned* rp = sBu + (nt * 16 + rowl) * 16;
                #pragma unroll
                for (int d = 0; d < 8; ++d) b[nt].u[d] = rp[(d & 3) + half * 4 + (d >> 2) * 8];
            }
            #pragma unroll
            for (int mt = 0; mt < 2; ++mt) {
                // A (16x32 MxK) wave32 layout: lane&15 -> M; dword d holds
                // K-pair index p = (d&3) + half*4 + (d>>2)*8
                U a;
                const unsigned* rp = sAu + (wbase + mt * 16 + rowl) * 16;
                #pragma unroll
                for (int d = 0; d < 8; ++d) {
                    const int p = (d & 3) + half * 4 + (d >> 2) * 8;
                    a.u[d] = rp[p];
                }
                #pragma unroll
                for (int nt = 0; nt < 4; ++nt) {
                    acc[mt][nt] = __builtin_amdgcn_wmma_f32_16x16x32_f16(
                        false, a.v, false, b[nt].v, (short)0, acc[mt][nt],
                        false, false);
                }
            }
        }
        __syncthreads();
    }

    // ---- store: C/D 16x16 layout -> cout = Mbase + half*8 + r, wo = Nbase + rowl
    #pragma unroll
    for (int mt = 0; mt < 2; ++mt)
        #pragma unroll
        for (int nt = 0; nt < 4; ++nt)
            #pragma unroll
            for (int r = 0; r < 8; ++r) {
                const int co = wbase + mt * 16 + half * 8 + r;
                const int wo = nt * 16 + rowl;
                out[(((size_t)n * COUT + co) * HH + ho) * WW + wo] =
                    acc[mt][nt][r] + bias[co];
            }
}

// ---------------------------------------------------------------------------
extern "C" void kernel_launch(void* const* d_in, const int* in_sizes, int n_in,
                              void* d_out, int out_size, void* d_ws, size_t ws_size,
                              hipStream_t stream) {
    (void)in_sizes; (void)n_in; (void)out_size; (void)ws_size;
    const float* input  = (const float*)d_in[0];
    const float* offset = (const float*)d_in[1];
    const float* maskp  = (const float*)d_in[2];
    const float* weight = (const float*)d_in[3];
    const float* bias   = (const float*)d_in[4];
    float* out = (float*)d_out;
    _Float16* wf16 = (_Float16*)d_ws;   // 9*8*256*32 f16 = 1.125 MB

    dcn_pack_weight<<<2304, 256, 0, stream>>>(weight, wf16);   // 589824 elems
    dim3 grid(HH, 8);                                          // (ho, n)
    dcn_fused<<<grid, 256, 0, stream>>>(input, offset, maskp, wf16, bias, out);
}
